// CSNMModule_38663295598699
// MI455X (gfx1250) — matrix operand order repaired and checked by reading, not compiled, not run on hardware
//
#include <hip/hip_runtime.h>

typedef __bf16 bf16_t;
typedef __attribute__((ext_vector_type(16))) __bf16 v16bf;
typedef __attribute__((ext_vector_type(8)))  float   v8f;

#define BATCH 8
#define NTOK  4096
#define DIM   512
#define K2    1024
#define RPB   128           // rows per GEMM block
#define LDA   1032          // A row stride in LDS (bf16 elems), 1024 + 8 pad
#define LDB   1032          // B col stride in LDS (bf16 elems)

#if defined(__gfx1250__) && __has_builtin(__builtin_amdgcn_global_load_async_to_lds_b128) && __has_builtin(__builtin_amdgcn_s_wait_asynccnt)
#define USE_ASYNC_LDS 1
#else
#define USE_ASYNC_LDS 0
#endif

#if USE_ASYNC_LDS
typedef int v4i_ __attribute__((vector_size(16)));
typedef __attribute__((address_space(1))) v4i_ gv4;
typedef __attribute__((address_space(3))) v4i_ lv4;
#endif

// ---------------- zero fsum accumulator ----------------
__global__ void k_zero(float* __restrict__ fsum) {
  int i = blockIdx.x * 256 + threadIdx.x;
  if (i < BATCH * DIM) fsum[i] = 0.f;
}

// ---------------- W1 -> bf16, transposed to [k][col][kk] ----------------
__global__ void k_w1bf(const float* __restrict__ W1, bf16_t* __restrict__ w1t) {
  int i = blockIdx.x * 256 + threadIdx.x;
  if (i < 3 * K2 * DIM) {
    int col = i & 511;
    int kk  = (i >> 9) & 1023;
    int k   = i >> 19;
    w1t[((size_t)k * DIM + col) * K2 + kk] = (bf16_t)W1[i];
  }
}

// ---------------- windows: 3x3x3 avg-pool in channel-mixed layout ----------------
// block = (s, b, c); output wbf[((s*8+b)*512+c)*4096 + g]  (pooled natural layout)
__global__ void __launch_bounds__(256)
k_window(const float* __restrict__ e0, const float* __restrict__ e1,
         const float* __restrict__ e2, bf16_t* __restrict__ wbf) {
  __shared__ float x[NTOK];
  int blk = blockIdx.x;
  int c  = blk & 511;
  int sb = blk >> 9;
  int b  = sb & 7;
  int s  = sb >> 3;
  const float* e = (s == 0) ? e0 : (s == 1) ? e1 : e2;
  const float* src = e + (size_t)b * NTOK * DIM + (size_t)c * NTOK;
#if USE_ASYNC_LDS
  {
    gv4* gs = (gv4*)src;
    lv4* xl = (lv4*)x;
#pragma unroll
    for (int i = 0; i < 4; ++i) {
      int idx = (int)threadIdx.x + i * 256;   // 16B units; 4096 floats = 1024 vecs
      __builtin_amdgcn_global_load_async_to_lds_b128(gs + idx, xl + idx, 0, 0);
    }
    __builtin_amdgcn_s_wait_asynccnt(0);
  }
#else
  for (int g = threadIdx.x; g < NTOK; g += 256) x[g] = src[g];
#endif
  __syncthreads();
  int dZ, dY, dX;
  if (s == 0)      { dZ = 16; dY = 16; dX = 16; }
  else if (s == 1) { dZ = 32; dY = 16; dX = 8;  }
  else             { dZ = 8;  dY = 32; dX = 16; }
  bf16_t* out = wbf + (size_t)blk * NTOK;
  for (int g = threadIdx.x; g < NTOK; g += 256) {
    int xx = g % dX; int rem = g / dX; int yy = rem % dY; int zz = rem / dY;
    float sum = 0.f;
    for (int dz = -1; dz <= 1; ++dz) {
      int z2 = zz + dz; if ((unsigned)z2 >= (unsigned)dZ) continue;
      for (int dy = -1; dy <= 1; ++dy) {
        int y2 = yy + dy; if ((unsigned)y2 >= (unsigned)dY) continue;
        int base = (z2 * dY + y2) * dX;
        for (int dx = -1; dx <= 1; ++dx) {
          int x2 = xx + dx; if ((unsigned)x2 >= (unsigned)dX) continue;
          sum += x[base + x2];
        }
      }
    }
    out[g] = (bf16_t)(sum * (1.0f / 27.0f));
  }
}

// ---------------- fused GEMM + relu + dot(W2): logits S[dir][b][n] ----------------
__global__ void __launch_bounds__(256)
k_gemm(const bf16_t* __restrict__ wbf, const bf16_t* __restrict__ w1t,
       const float* __restrict__ b1, const float* __restrict__ W2,
       const float* __restrict__ b2, float* __restrict__ S) {
  extern __shared__ char smem[];
  bf16_t* As = (bf16_t*)smem;            // [RPB][LDA]
  bf16_t* Bt = As + RPB * LDA;           // [16][LDB] (col-major tile of W1)

  int blk = blockIdx.x;
  int rb  = blk & 31;
  int t   = blk >> 5;
  int b   = t & 7;
  int dir = t >> 3;
  int k   = dir >> 1;
  constexpr int PI[3] = {0, 0, 1};
  constexpr int PJ[3] = {1, 2, 2};
  int winFirst  = (dir & 1) ? PJ[k] : PI[k];
  int winSecond = (dir & 1) ? PI[k] : PJ[k];
  int n0  = rb * RPB;
  int tid = threadIdx.x;

  // Stage A: As[r][kk] = window[first|second][b][kk%512][n0+r]  (LDS does the transpose)
  // One 8-byte global load (4 rows along contiguous g) + 4 ds b16 stores.
  const bf16_t* wfp = wbf + (size_t)(winFirst  * BATCH + b) * DIM * NTOK;
  const bf16_t* wsp = wbf + (size_t)(winSecond * BATCH + b) * DIM * NTOK;
  for (int idx = tid; idx < RPB * K2 / 4; idx += 256) {
    int r4 = (idx & 31) * 4;
    int kk = idx >> 5;
    const bf16_t* srcrow = (kk < DIM) ? (wfp + (size_t)kk * NTOK)
                                      : (wsp + (size_t)(kk - DIM) * NTOK);
    union { unsigned long long u; bf16_t h[4]; } uu;
    uu.u = *(const unsigned long long*)(srcrow + n0 + r4);
#pragma unroll
    for (int i = 0; i < 4; ++i) As[(r4 + i) * LDA + kk] = uu.h[i];
  }

  const bf16_t* w1k = w1t + (size_t)k * DIM * K2;   // [col][kk]
  int lane  = tid & 31;
  int wave  = tid >> 5;
  int laneM = lane & 15;
  int laneH = lane >> 4;
  int rloc  = wave * 16 + laneM;
  float b2v = b2[k];

  float rowAcc[8];
#pragma unroll
  for (int v = 0; v < 8; ++v) rowAcc[v] = 0.f;

  for (int jt = 0; jt < DIM / 16; ++jt) {
    __syncthreads();
    // Stage B tile: Bt[col][kk] = w1t[k][jt*16+col][kk]  (contiguous rows -> async DMA)
#if USE_ASYNC_LDS
    {
      const bf16_t* w1col0 = w1k + (size_t)jt * 16 * K2;
#pragma unroll
      for (int i = 0; i < 8; ++i) {
        int idx = tid + i * 256;        // 16 cols x 128 chunks of 16B
        int col = idx >> 7;
        int ch  = idx & 127;
        gv4* gs = (gv4*)(w1col0 + (size_t)col * K2) + ch;
        lv4* ld = (lv4*)(Bt + col * LDB) + ch;
        __builtin_amdgcn_global_load_async_to_lds_b128(gs, ld, 0, 0);
      }
      __builtin_amdgcn_s_wait_asynccnt(0);
    }
#else
    for (int idx = tid; idx < 16 * K2; idx += 256) {
      int col = idx >> 10;
      int kk  = idx & 1023;
      Bt[col * LDB + kk] = w1k[(size_t)(jt * 16 + col) * K2 + kk];
    }
#endif
    __syncthreads();

    union { v8f v; float f[8]; } acc0, acc1;
#pragma unroll
    for (int z = 0; z < 8; ++z) { acc0.f[z] = 0.f; acc1.f[z] = 0.f; }

    union U16 { v16bf v; uint4 q[2]; };
    const char* arow = (const char*)As + (size_t)rloc * LDA * 2;
    const char* bcol = (const char*)Bt + (size_t)laneM * LDB * 2;
    // Two independent accumulator chains (even/odd K-steps) for WMMA ILP.
#pragma unroll 2
    for (int ks = 0; ks < K2 / 32; ks += 2) {
      U16 Af, Bf;
      // A frag (16x32 bf16): lanes 0-15 rows M=lane, K runs {0..7,16..23}; lanes 16-31 {8..15,24..31}
      int abyte = ks * 64 + laneH * 16;
      Af.q[0] = *(const uint4*)(arow + abyte);
      Af.q[1] = *(const uint4*)(arow + abyte + 32);
      // B frag (32x16 bf16): lane col N=lane&15; lanes 0-15 K {0..7,16..23}; lanes 16-31 {8..15,24..31}
      int bbyte = ks * 64 + laneH * 16;
      Bf.q[0] = *(const uint4*)(bcol + bbyte);
      Bf.q[1] = *(const uint4*)(bcol + bbyte + 32);
      acc0.v = __builtin_amdgcn_wmma_f32_16x16x32_bf16(false, Af.v, false, Bf.v,
                                                       (short)0, acc0.v, false, false);
      U16 Ag, Bg;
      Ag.q[0] = *(const uint4*)(arow + abyte + 64);
      Ag.q[1] = *(const uint4*)(arow + abyte + 96);
      Bg.q[0] = *(const uint4*)(bcol + bbyte + 64);
      Bg.q[1] = *(const uint4*)(bcol + bbyte + 96);
      acc1.v = __builtin_amdgcn_wmma_f32_16x16x32_bf16(false, Ag.v, false, Bg.v,
                                                       (short)0, acc1.v, false, false);
    }

    // C layout: lane col = lane&15, rows = laneH*8 + v.
    // Branchless relu * W2, then reduce the 16 lanes sharing each row (xor masks
    // 1/2/4/8 stay within the laneH half). Row partials accumulate in registers.
    int j = jt * 16 + laneM;
    float b1v = b1[k * DIM + j];
    float w2v = W2[k * DIM + j];
#pragma unroll
    for (int v = 0; v < 8; ++v) {
      float val = fmaxf(acc0.f[v] + acc1.f[v] + b1v, 0.f) * w2v;
      val += __shfl_xor(val, 1, 32);
      val += __shfl_xor(val, 2, 32);
      val += __shfl_xor(val, 4, 32);
      val += __shfl_xor(val, 8, 32);
      rowAcc[v] += val;   // valid on laneM==0 lanes
    }
  }

  if (laneM == 0) {
    size_t sbase = (size_t)(dir * BATCH + b) * NTOK + n0 + wave * 16 + laneH * 8;
#pragma unroll
    for (int v = 0; v < 8; ++v) S[sbase + v] = rowAcc[v] + b2v;
  }
}

// ---------------- softmax over N + weighted sum of embed -> matched[dir][b][d] ----------------
__global__ void __launch_bounds__(256)
k_match(const float* __restrict__ S, const float* __restrict__ e0,
        const float* __restrict__ e1, const float* __restrict__ e2,
        float* __restrict__ matched) {
  __shared__ float p[NTOK];
  __shared__ float red[256];
  int blk = blockIdx.x;
  int dh  = blk & 1;
  int t   = blk >> 1;
  int b   = t & 7;
  int dir = t >> 3;
  constexpr int srcIdx[6] = {1, 0, 2, 0, 2, 1};
  int si = srcIdx[dir];
  const float* e = (si == 0) ? e0 : (si == 1) ? e1 : e2;
  const float* s = S + (size_t)(dir * BATCH + b) * NTOK;
  int tid = threadIdx.x;

  float mx = -1e30f;
  for (int n = tid; n < NTOK; n += 256) mx = fmaxf(mx, s[n]);
  red[tid] = mx; __syncthreads();
  for (int o = 128; o; o >>= 1) { if (tid < o) red[tid] = fmaxf(red[tid], red[tid + o]); __syncthreads(); }
  mx = red[0]; __syncthreads();

  float sum = 0.f;
  for (int n = tid; n < NTOK; n += 256) { float v = expf(s[n] - mx); p[n] = v; sum += v; }
  red[tid] = sum; __syncthreads();
  for (int o = 128; o; o >>= 1) { if (tid < o) red[tid] += red[tid + o]; __syncthreads(); }
  float inv = 1.f / red[0];
  __syncthreads();

  int d = dh * 256 + tid;
  const float* eb = e + (size_t)b * NTOK * DIM + d;
  float acc = 0.f;
  for (int n = 0; n < NTOK; ++n) {
    __builtin_prefetch(eb + (size_t)(n + 16) * DIM, 0, 1);  // global_prefetch_b8
    acc += p[n] * eb[(size_t)n * DIM];
  }
  matched[(size_t)(dir * BATCH + b) * DIM + d] = acc * inv;
}

// ---------------- sum of all embed rows -> fsum[b][d] ----------------
__global__ void __launch_bounds__(256)
k_sumemb(const float* __restrict__ e0, const float* __restrict__ e1,
         const float* __restrict__ e2, float* __restrict__ fsum) {
  int blk = blockIdx.x;
  int nc = blk & 31;
  int b  = blk >> 5;
  int tid = threadIdx.x;
  float a0 = 0.f, a1 = 0.f;
  const size_t base = (size_t)b * NTOK * DIM + (size_t)nc * 128 * DIM;
  for (int n = 0; n < 128; ++n) {
    size_t o = base + (size_t)n * DIM;
    a0 += e0[o + tid]       + e1[o + tid]       + e2[o + tid];
    a1 += e0[o + tid + 256] + e1[o + tid + 256] + e2[o + tid + 256];
  }
  atomicAdd(&fsum[b * DIM + tid],       a0);
  atomicAdd(&fsum[b * DIM + tid + 256], a1);
}

// ---------------- fused mean + LayerNorm + @Wf + bf ----------------
__global__ void __launch_bounds__(256)
k_final(const float* __restrict__ fsum, const float* __restrict__ matched,
        const float* __restrict__ gamma, const float* __restrict__ beta,
        const float* __restrict__ Wf, const float* __restrict__ bfv,
        float* __restrict__ out) {
  __shared__ float ln[DIM];
  __shared__ float red[256];
  int b = blockIdx.x;
  int tid = threadIdx.x;
  float f0 = fsum[b * DIM + tid], f1 = fsum[b * DIM + tid + 256];
  for (int dir = 0; dir < 6; ++dir) {
    const float* m = matched + (size_t)(dir * BATCH + b) * DIM;
    f0 += m[tid]; f1 += m[tid + 256];
  }
  const float invR = 1.f / 12294.f;   // 3*4096 + 6 rows
  f0 *= invR; f1 *= invR;
  red[tid] = f0 + f1; __syncthreads();
  for (int o = 128; o; o >>= 1) { if (tid < o) red[tid] += red[tid + o]; __syncthreads(); }
  float mu = red[0] * (1.f / DIM); __syncthreads();
  float d0 = f0 - mu, d1 = f1 - mu;
  red[tid] = d0 * d0 + d1 * d1; __syncthreads();
  for (int o = 128; o; o >>= 1) { if (tid < o) red[tid] += red[tid + o]; __syncthreads(); }
  float rstd = rsqrtf(red[0] * (1.f / DIM) + 1e-5f);
  ln[tid]       = d0 * rstd * gamma[tid]       + beta[tid];
  ln[tid + 256] = d1 * rstd * gamma[tid + 256] + beta[tid + 256];
  __syncthreads();
  for (int oo = tid; oo < DIM; oo += 256) {
    float acc = bfv[oo];
    for (int dd = 0; dd < DIM; ++dd) acc += ln[dd] * Wf[(size_t)dd * DIM + oo];
    out[(size_t)b * DIM + oo] = acc;
  }
}

extern "C" void kernel_launch(void* const* d_in, const int* in_sizes, int n_in,
                              void* d_out, int out_size, void* d_ws, size_t ws_size,
                              hipStream_t stream) {
  const float* e0 = (const float*)d_in[0];
  const float* e1 = (const float*)d_in[1];
  const float* e2 = (const float*)d_in[2];
  const float* W1 = (const float*)d_in[3];
  const float* b1 = (const float*)d_in[4];
  const float* W2 = (const float*)d_in[5];
  const float* b2 = (const float*)d_in[6];
  const float* gamma = (const float*)d_in[7];
  const float* beta  = (const float*)d_in[8];
  const float* Wf = (const float*)d_in[9];
  const float* bf = (const float*)d_in[10];
  float* out = (float*)d_out;

  // workspace layout (bytes)
  char* ws = (char*)d_ws;
  bf16_t* wbf     = (bf16_t*)ws;                                  // 3*8*512*4096*2 = 100,663,296
  bf16_t* w1t     = (bf16_t*)(ws + 100663296);                    // 3*512*1024*2   =   3,145,728
  float*  Slog    = (float*)(ws + 100663296 + 3145728);           // 6*8*4096*4     =     786,432
  float*  matched = (float*)(ws + 100663296 + 3145728 + 786432);  // 6*8*512*4      =      98,304
  float*  fsum    = (float*)(ws + 100663296 + 3145728 + 786432 + 98304); // 8*512*4 =      16,384

  k_zero  <<<16,    256, 0, stream>>>(fsum);
  k_w1bf  <<<6144,  256, 0, stream>>>(W1, w1t);
  k_window<<<12288, 256, 0, stream>>>(e0, e1, e2, wbf);

  size_t smem = (size_t)(RPB * LDA + 16 * LDB) * sizeof(bf16_t);
  k_gemm  <<<1536,  256, smem, stream>>>(wbf, w1t, b1, W2, b2, Slog);

  k_match <<<96,    256, 0, stream>>>(Slog, e0, e1, e2, matched);
  k_sumemb<<<256,   256, 0, stream>>>(e0, e1, e2, fsum);
  k_final <<<8,     256, 0, stream>>>(fsum, matched, gamma, beta, Wf, bf, out);
}
